// GatedMultiheadAttention_84859963834492
// MI455X (gfx1250) — matrix operand order repaired and checked by reading, not compiled, not run on hardware
//
#include <hip/hip_runtime.h>
#include <hip/hip_bf16.h>

// ---------------------------------------------------------------------------
// Gated multi-head attention for MI455X (gfx1250), bf16 WMMA pipeline.
//   out  = sigmoid((xWq^T)(xWk^T)^T / sqrt(HD)) (xWv^T)  Wo^T  (+ biases)
//   d_out = [ out (B,L,D) f32 | attn_weights (B,H,L,S) f32 ]
// Workspace requirement: 64 MB.
// ---------------------------------------------------------------------------

typedef __bf16 bf16_t;
typedef __attribute__((ext_vector_type(16))) __bf16 v16bf;
typedef __attribute__((ext_vector_type(8)))  __bf16 v8bf;
typedef __attribute__((ext_vector_type(8)))  float  v8f;

#define B_  2
#define L_  2048
#define S_  2048
#define D_  1024
#define H_  16
#define HD_ 64

// ---------------------------------------------------------------------------
// WMMA fragment loaders (CDNA5 ISA 7.12.2, wave32):
//  A (16x32 bf16): lane l holds row m=l&15; elems 0..7 -> K = 8*hi + j,
//                  elems 8..15 -> K = 16 + 8*hi + j   (hi = l>>4)
//  B (32x16 bf16) loaded from a row-major "transposed" operand T where
//                  B[k][n] = T[n][k]: lane l holds col n=l&15;
//                  elems 0..7 -> K = 8*hi + j, elems 8..15 -> K = 16 + 8*hi + j.
//  C/D (16x16 f32): lane l col n=l&15; elem r -> row m = r + 8*hi.
// ---------------------------------------------------------------------------
__device__ __forceinline__ v16bf load_a_frag(const bf16_t* base, int stride, int lane) {
    const int m  = lane & 15;
    const int hi = lane >> 4;
    const bf16_t* p = base + (size_t)m * stride + 8 * hi;
    union { v16bf v; v8bf h[2]; } u;
    u.h[0] = *(const v8bf*)(p);
    u.h[1] = *(const v8bf*)(p + 16);
    return u.v;
}

__device__ __forceinline__ v16bf load_bT_frag(const bf16_t* base, int stride, int lane) {
    const int n  = lane & 15;
    const int hi = lane >> 4;
    const bf16_t* p = base + (size_t)n * stride + 8 * hi;
    union { v16bf v; v8bf h[2]; } u;
    u.h[0] = *(const v8bf*)(p);
    u.h[1] = *(const v8bf*)(p + 16);
    return u.v;
}

// ---------------------------------------------------------------------------
// f32 -> bf16 conversion
// ---------------------------------------------------------------------------
__global__ __launch_bounds__(256) void f32_to_bf16_kernel(
    const float* __restrict__ in, bf16_t* __restrict__ out, int n) {
    int i = blockIdx.x * 256 + threadIdx.x;
    if (i < n) out[i] = (bf16_t)in[i];
}

// ---------------------------------------------------------------------------
// Generic NT GEMM: C[m][n] = sum_k A[m][k] * W[n][k] + bias[n]
//   A: (4096 x 1024) bf16 row-major, W: (1024 x 1024) bf16 row-major.
//   One wave computes a 64x32 tile; manual ping-pong double buffering
//   (two named fragment buffers, no rotation copies). 8 waves / block.
//   mode 0: bf16 out, (B,H,T,HD) head-split layout      (Q, K)
//   mode 1: bf16 out, (B,H,HD,S) head-split transposed  (V)
//   mode 2: f32 out (nontemporal), row-major (B*L, D)   (final projection)
// ---------------------------------------------------------------------------
struct Frags { v16bf a[4]; v16bf b[2]; };

__device__ __forceinline__ void load_tile(Frags& f, const bf16_t* const* Arow,
                                          const bf16_t* const* Wrow, int k, int lane) {
#pragma unroll
    for (int i = 0; i < 4; ++i) f.a[i] = load_a_frag (Arow[i] + k, D_, lane);
#pragma unroll
    for (int j = 0; j < 2; ++j) f.b[j] = load_bT_frag(Wrow[j] + k, D_, lane);
}

__device__ __forceinline__ void mma_tile(v8f cc[4][2], const Frags& f) {
#pragma unroll
    for (int i = 0; i < 4; ++i)
#pragma unroll
        for (int j = 0; j < 2; ++j)
            cc[i][j] = __builtin_amdgcn_wmma_f32_16x16x32_bf16(
                false, f.a[i], false, f.b[j], (short)0, cc[i][j], false, false);
}

__global__ __launch_bounds__(256) void gemm_bf16_kernel(
    const bf16_t* __restrict__ A, const bf16_t* __restrict__ W,
    const float* __restrict__ bias, void* __restrict__ Out, int mode)
{
    const int lane = threadIdx.x & 31;
    const int task = blockIdx.x * 8 + (threadIdx.x >> 5);
    const int n32c = D_ / 32;                  // 32 column tiles
    const int m0 = (task / n32c) * 64;
    const int n0 = (task % n32c) * 32;

    const bf16_t* Arow[4];
    const bf16_t* Wrow[2];
#pragma unroll
    for (int i = 0; i < 4; ++i) Arow[i] = A + (size_t)(m0 + 16 * i) * D_;
#pragma unroll
    for (int j = 0; j < 2; ++j) Wrow[j] = W + (size_t)(n0 + 16 * j) * D_;

    v8f cc[4][2] = {};
    Frags cur, nxt;
    load_tile(cur, Arow, Wrow, 0, lane);

    int k0 = 0;
    for (; k0 < D_ - 64; k0 += 64) {           // ping-pong over two k-tiles
        load_tile(nxt, Arow, Wrow, k0 + 32, lane);
        const int kp = (k0 + 128 < D_) ? (k0 + 128) : (D_ - 32);
        __builtin_prefetch(Arow[0] + kp, 0, 3);
        __builtin_prefetch(Wrow[0] + kp, 0, 3);
        mma_tile(cc, cur);
        load_tile(cur, Arow, Wrow, k0 + 64, lane);
        mma_tile(cc, nxt);
    }
    load_tile(nxt, Arow, Wrow, D_ - 32, lane); // last tile
    mma_tile(cc, cur);
    mma_tile(cc, nxt);

    const int nloc = lane & 15;
    const int hi   = lane >> 4;
#pragma unroll
    for (int sm = 0; sm < 4; ++sm) {
#pragma unroll
        for (int sn = 0; sn < 2; ++sn) {
            const int n  = n0 + sn * 16 + nloc;
            const float bn = bias[n];
#pragma unroll
            for (int r = 0; r < 8; ++r) {
                const int m = m0 + sm * 16 + 8 * hi + r;
                const float val = cc[sm][sn][r] + bn;
                if (mode == 2) {
                    __builtin_nontemporal_store(val, (float*)Out + (size_t)m * D_ + n);
                } else {
                    const int b_ = m >> 11, t = m & (L_ - 1);   // L_ == 2048
                    const int h = n >> 6, hd = n & (HD_ - 1);
                    const bf16_t bvv = (bf16_t)val;
                    if (mode == 0)
                        ((bf16_t*)Out)[(((size_t)(b_ * H_ + h)) * L_ + t) * HD_ + hd] = bvv;
                    else
                        ((bf16_t*)Out)[(((size_t)(b_ * H_ + h)) * HD_ + hd) * S_ + t] = bvv;
                }
            }
        }
    }
}

// ---------------------------------------------------------------------------
// Fused attention: one wave per (b, h, 16-row L tile).
//   scores = Q Khead^T / 8  -> mask -> sigmoid -> stream f32 attn tile (NT),
//   re-stripe tile via LDS into A-fragment layout, accumulate O = attn * V.
// ---------------------------------------------------------------------------
__global__ __launch_bounds__(256) void attn_kernel(
    const bf16_t* __restrict__ Qh, const bf16_t* __restrict__ Kh,
    const bf16_t* __restrict__ Vt, const unsigned char* __restrict__ mask,
    float* __restrict__ attnOut, bf16_t* __restrict__ Obuf)
{
    __shared__ bf16_t smem[8][16 * 32];        // 1 KB per wave
    const int lane = threadIdx.x & 31;
    const int wid  = threadIdx.x >> 5;
    const int task = blockIdx.x * 8 + wid;
    const int ltiles = L_ / 16;                // 128
    const int bh = task / ltiles;              // b*H + h
    const int l0 = (task % ltiles) * 16;
    const int b  = bh / H_;
    const int h  = bh % H_;

    const bf16_t* Qbase = Qh + ((size_t)bh * L_ + l0) * HD_;
    const bf16_t* Kbase = Kh + (size_t)bh * S_ * HD_;
    const bf16_t* Vbase = Vt + (size_t)bh * HD_ * S_;
    float* attnBase = attnOut + ((size_t)bh * L_ + l0) * S_;

    // Q A-fragments over K = 0..31, 32..63 (loaded once)
    const v16bf qa0 = load_a_frag(Qbase +  0, HD_, lane);
    const v16bf qa1 = load_a_frag(Qbase + 32, HD_, lane);

    v8f oacc[4] = {};
    bf16_t* my = smem[wid];
    const int nloc = lane & 15;
    const int hi   = lane >> 4;

    for (int s0 = 0; s0 < S_; s0 += 32) {
#pragma unroll
        for (int t = 0; t < 2; ++t) {
            const int scol = s0 + t * 16;
            v16bf kb0 = load_bT_frag(Kbase + (size_t)scol * HD_ +  0, HD_, lane);
            v16bf kb1 = load_bT_frag(Kbase + (size_t)scol * HD_ + 32, HD_, lane);
            v8f c = {};
            c = __builtin_amdgcn_wmma_f32_16x16x32_bf16(false, qa0, false, kb0, (short)0, c, false, false);
            c = __builtin_amdgcn_wmma_f32_16x16x32_bf16(false, qa1, false, kb1, (short)0, c, false, false);

            const int s = scol + nloc;
            const bool msk = mask[(size_t)b * S_ + s] != 0;
#pragma unroll
            for (int r = 0; r < 8; ++r) {
                const float sc = c[r] * 0.125f;                     // 1/sqrt(64)
                const float w  = msk ? 0.0f : 1.0f / (1.0f + __expf(-sc));
                const int m = 8 * hi + r;
                // 537 MB write-once stream: keep it out of L2 (TH=NT)
                __builtin_nontemporal_store(w, attnBase + (size_t)m * S_ + s);
                my[m * 32 + t * 16 + nloc] = (bf16_t)w;             // LDS stage
            }
        }
        asm volatile("" ::: "memory");  // keep LDS stores before LDS loads
        // attn A-fragment from LDS (per-wave in-order DS guarantees visibility)
        v16bf aa;
        {
            const bf16_t* p = my + nloc * 32 + 8 * hi;
            union { v16bf v; v8bf hh[2]; } u;
            u.hh[0] = *(const v8bf*)(p);
            u.hh[1] = *(const v8bf*)(p + 16);
            aa = u.v;
        }
        asm volatile("" ::: "memory");
        // V B-fragments for d-tiles 0,16,32,48 (Vt is (HD,S) row-major)
        v16bf v0 = load_bT_frag(Vbase + (size_t) 0 * S_ + s0, S_, lane);
        v16bf v1 = load_bT_frag(Vbase + (size_t)16 * S_ + s0, S_, lane);
        v16bf v2 = load_bT_frag(Vbase + (size_t)32 * S_ + s0, S_, lane);
        v16bf v3 = load_bT_frag(Vbase + (size_t)48 * S_ + s0, S_, lane);
        oacc[0] = __builtin_amdgcn_wmma_f32_16x16x32_bf16(false, aa, false, v0, (short)0, oacc[0], false, false);
        oacc[1] = __builtin_amdgcn_wmma_f32_16x16x32_bf16(false, aa, false, v1, (short)0, oacc[1], false, false);
        oacc[2] = __builtin_amdgcn_wmma_f32_16x16x32_bf16(false, aa, false, v2, (short)0, oacc[2], false, false);
        oacc[3] = __builtin_amdgcn_wmma_f32_16x16x32_bf16(false, aa, false, v3, (short)0, oacc[3], false, false);
    }

    // Write per-head output back in (B*L, D) row-major bf16 for the Wo GEMM
#pragma unroll
    for (int sn = 0; sn < 4; ++sn) {
        const int d = sn * 16 + nloc;
#pragma unroll
        for (int r = 0; r < 8; ++r) {
            const int m = 8 * hi + r;
            Obuf[((size_t)(b * L_ + l0 + m)) * D_ + h * HD_ + d] = (bf16_t)oacc[sn][r];
        }
    }
}

// ---------------------------------------------------------------------------
// Host-side launch
// ---------------------------------------------------------------------------
extern "C" void kernel_launch(void* const* d_in, const int* in_sizes, int n_in,
                              void* d_out, int out_size, void* d_ws, size_t ws_size,
                              hipStream_t stream) {
    (void)in_sizes; (void)n_in; (void)out_size; (void)ws_size;

    const float* q    = (const float*)d_in[0];
    const float* k    = (const float*)d_in[1];
    const float* v    = (const float*)d_in[2];
    const unsigned char* mask = (const unsigned char*)d_in[3];
    const float* Wq = (const float*)d_in[4];
    const float* bq = (const float*)d_in[5];
    const float* Wk = (const float*)d_in[6];
    const float* bk = (const float*)d_in[7];
    const float* Wv = (const float*)d_in[8];
    const float* bv = (const float*)d_in[9];
    const float* Wo = (const float*)d_in[10];
    const float* bo = (const float*)d_in[11];

    const size_t MB = 1u << 20;
    char* ws = (char*)d_ws;
    bf16_t* Wq_bf = (bf16_t*)(ws + 0 * MB);
    bf16_t* Wk_bf = (bf16_t*)(ws + 2 * MB);
    bf16_t* Wv_bf = (bf16_t*)(ws + 4 * MB);
    bf16_t* Wo_bf = (bf16_t*)(ws + 6 * MB);
    bf16_t* Xq_bf = (bf16_t*)(ws + 8 * MB);
    bf16_t* Xk_bf = (bf16_t*)(ws + 16 * MB);
    bf16_t* Xv_bf = (bf16_t*)(ws + 24 * MB);
    bf16_t* Qh    = (bf16_t*)(ws + 32 * MB);
    bf16_t* Kh    = (bf16_t*)(ws + 40 * MB);
    bf16_t* Vt    = (bf16_t*)(ws + 48 * MB);
    bf16_t* Obuf  = (bf16_t*)(ws + 56 * MB);   // total 64 MB

    const int DD  = D_ * D_;          // 1,048,576
    const int BLD = B_ * L_ * D_;     // 4,194,304

    f32_to_bf16_kernel<<<(DD  + 255) / 256, 256, 0, stream>>>(Wq, Wq_bf, DD);
    f32_to_bf16_kernel<<<(DD  + 255) / 256, 256, 0, stream>>>(Wk, Wk_bf, DD);
    f32_to_bf16_kernel<<<(DD  + 255) / 256, 256, 0, stream>>>(Wv, Wv_bf, DD);
    f32_to_bf16_kernel<<<(DD  + 255) / 256, 256, 0, stream>>>(Wo, Wo_bf, DD);
    f32_to_bf16_kernel<<<(BLD + 255) / 256, 256, 0, stream>>>(q, Xq_bf, BLD);
    f32_to_bf16_kernel<<<(BLD + 255) / 256, 256, 0, stream>>>(k, Xk_bf, BLD);
    f32_to_bf16_kernel<<<(BLD + 255) / 256, 256, 0, stream>>>(v, Xv_bf, BLD);

    // GEMM: 64x32 tiles -> (4096/64)*(1024/32) = 2048 wave-tasks, 8/block
    gemm_bf16_kernel<<<256, 256, 0, stream>>>(Xq_bf, Wq_bf, bq, Qh, 0);
    gemm_bf16_kernel<<<256, 256, 0, stream>>>(Xk_bf, Wk_bf, bk, Kh, 0);
    gemm_bf16_kernel<<<256, 256, 0, stream>>>(Xv_bf, Wv_bf, bv, Vt, 1);

    float* outMain = (float*)d_out;
    float* attnOut = outMain + (size_t)BLD;    // tuple part 2: (B,H,L,S)
    attn_kernel<<<512, 256, 0, stream>>>(Qh, Kh, Vt, mask, attnOut, Obuf);

    gemm_bf16_kernel<<<256, 256, 0, stream>>>(Obuf, Wo_bf, bo, outMain, 2);
}
